// ContrastiveLoss_70085276336527
// MI455X (gfx1250) — compile-verified, matrix operand order checked by reading, not run on hardware
//
#include <hip/hip_runtime.h>
#include <math.h>

#define B_  16
#define T_  4096
#define D_  256
#define TEMP_ 0.07f
#define EPS_  1e-8f

typedef __attribute__((ext_vector_type(16))) _Float16 v16h;
typedef __attribute__((ext_vector_type(8)))  _Float16 v8h;
typedef __attribute__((ext_vector_type(8)))  float    v8f;

// ---------------------------------------------------------------------------
// Kernel 1: idx[b] = argmax(mask[b,:]) with first-occurrence tie-break
// ---------------------------------------------------------------------------
__global__ __launch_bounds__(256) void argmax_kernel(const int* __restrict__ mask,
                                                     int* __restrict__ idx) {
    const int b = blockIdx.x;
    const int* row = mask + (size_t)b * T_;
    int bestV = -2147483647, bestI = T_;
    for (int t = threadIdx.x; t < T_; t += 256) {
        int v = row[t];
        if (v > bestV) { bestV = v; bestI = t; }   // ascending t -> earliest kept
    }
    __shared__ int sv[256], si[256];
    sv[threadIdx.x] = bestV; si[threadIdx.x] = bestI;
    __syncthreads();
    for (int s = 128; s > 0; s >>= 1) {
        if (threadIdx.x < s) {
            int v2 = sv[threadIdx.x + s], i2 = si[threadIdx.x + s];
            if (v2 > sv[threadIdx.x] ||
                (v2 == sv[threadIdx.x] && i2 < si[threadIdx.x])) {
                sv[threadIdx.x] = v2; si[threadIdx.x] = i2;
            }
        }
        __syncthreads();
    }
    if (threadIdx.x == 0) idx[b] = si[0];
}

// ---------------------------------------------------------------------------
// Kernel 2: normalized anchors (f16 copies), pos[b], zero accumulators
// ---------------------------------------------------------------------------
__device__ __forceinline__ float blockSum256(float x) {
    __shared__ float s[256];
    const int t = threadIdx.x;
    s[t] = x; __syncthreads();
    for (int k = 128; k > 0; k >>= 1) { if (t < k) s[t] += s[t + k]; __syncthreads(); }
    float r = s[0]; __syncthreads();
    return r;
}

__global__ __launch_bounds__(256) void anchor_kernel(const float* __restrict__ video,
                                                     const float* __restrict__ audio,
                                                     const int* __restrict__ idx,
                                                     float* __restrict__ pos,
                                                     float* __restrict__ negA,
                                                     float* __restrict__ negV,
                                                     _Float16* __restrict__ vanch,
                                                     _Float16* __restrict__ aanch) {
    const int b = blockIdx.x;
    const int d = threadIdx.x;
    const int t = idx[b];
    const size_t base = ((size_t)(b * T_ + t)) * D_;
    const float v = video[base + d];
    const float a = audio[base + d];
    const float nv = blockSum256(v * v);
    const float na = blockSum256(a * a);
    const float vn = v / fmaxf(sqrtf(nv), EPS_);
    const float an = a / fmaxf(sqrtf(na), EPS_);
    vanch[b * D_ + d] = (_Float16)vn;
    aanch[b * D_ + d] = (_Float16)an;
    const float p = blockSum256(vn * an);
    if (d == 0) {
        pos[b]  = p / TEMP_;
        negA[b] = 0.f;   // a2v accumulator
        negV[b] = 0.f;   // v2a accumulator
    }
}

// ---------------------------------------------------------------------------
// Kernel 3: streaming WMMA pass.  One wave = one 16-row tile of T, processing
// both video (vs audio anchor) and audio (vs video anchor).
//
// A fragment (16x32 f16): lane m=lane&15 holds row m; per-lane K offsets are
//   koff = (lane<16) ? 0 : 8;  elems[0..7] = K(kc*32+koff+i), elems[8..15] = +16
// B fragment: anchor replicated across all 16 columns -> identical per-lane
//   K addressing, so every output lane carries a full sim column.
// ---------------------------------------------------------------------------
__device__ __forceinline__ v16h loadA16(const float* __restrict__ p0, float& nsum) {
    const float4 a = *(const float4*)(p0 + 0);
    const float4 b = *(const float4*)(p0 + 4);
    const float4 c = *(const float4*)(p0 + 16);
    const float4 d = *(const float4*)(p0 + 20);
    float f[16] = { a.x,a.y,a.z,a.w, b.x,b.y,b.z,b.w,
                    c.x,c.y,c.z,c.w, d.x,d.y,d.z,d.w };
    v16h r;
#pragma unroll
    for (int i = 0; i < 16; ++i) { nsum += f[i] * f[i]; r[i] = (_Float16)f[i]; }
    return r;
}

__device__ __forceinline__ v16h loadB16(const _Float16* __restrict__ p0) {
    const v8h lo = *(const v8h*)(p0 + 0);
    const v8h hi = *(const v8h*)(p0 + 16);
    v16h r;
#pragma unroll
    for (int i = 0; i < 8; ++i) { r[i] = lo[i]; r[i + 8] = hi[i]; }
    return r;
}

__global__ __launch_bounds__(128) void sim_kernel(const float* __restrict__ video,
                                                  const float* __restrict__ audio,
                                                  const int* __restrict__ idx,
                                                  const _Float16* __restrict__ aanch,
                                                  const _Float16* __restrict__ vanch,
                                                  float* __restrict__ negA,
                                                  float* __restrict__ negV) {
    const int b    = blockIdx.y;
    const int wave = threadIdx.x >> 5;
    const int lane = threadIdx.x & 31;
    const int tile = blockIdx.x * 4 + wave;          // 0..255
    const int t0   = tile * 16;
    const int m    = lane & 15;                      // row this lane feeds
    const int koff = (lane < 16) ? 0 : 8;            // per-lane K offset (CDNA5 layout)

    const float*    vrow = video + ((size_t)(b * T_ + t0 + m)) * D_;
    const float*    arow = audio + ((size_t)(b * T_ + t0 + m)) * D_;
    const _Float16* aF   = aanch + b * D_;           // a_anc -> multiplies video rows
    const _Float16* vF   = vanch + b * D_;           // v_anc -> multiplies audio rows

    v8f accV = {}; v8f accA = {};
    float nsumV = 0.f, nsumA = 0.f;

#pragma unroll
    for (int kc = 0; kc < 8; ++kc) {                 // D_/32 = 8 K-chunks
        const int kb = kc * 32 + koff;
        if (kc < 7) {                                // uniform branch; prefetch next chunk
            __builtin_prefetch(vrow + kb + 32, 0, 0);
            __builtin_prefetch(arow + kb + 32, 0, 0);
        }
        v16h av = loadA16(vrow + kb, nsumV);
        v16h aa = loadA16(arow + kb, nsumA);
        v16h bA = loadB16(aF + kb);
        v16h bV = loadB16(vF + kb);
        accV = __builtin_amdgcn_wmma_f32_16x16x32_f16(false, av, false, bA,
                                                      (short)0, accV, false, false);
        accA = __builtin_amdgcn_wmma_f32_16x16x32_f16(false, aa, false, bV,
                                                      (short)0, accA, false, false);
    }

    // lanes m and m+16 each accumulated disjoint halves of row m's sum-of-squares
    nsumV += __shfl_xor(nsumV, 16);
    nsumA += __shfl_xor(nsumA, 16);

    // D layout: VGPR r -> row r (lanes 0-15) / row r+8 (lanes 16-31)
    const int rbase = (lane < 16) ? 0 : 8;
    const int idxb  = idx[b];
    float sumV = 0.f, sumA = 0.f;
#pragma unroll
    for (int r = 0; r < 8; ++r) {
        const int trow = t0 + rbase + r;
        const float n2v = __shfl(nsumV, rbase + r);  // norm^2 of row lives in lane (row)
        const float n2a = __shfl(nsumA, rbase + r);
        const float sV  = accV[r] / fmaxf(sqrtf(n2v), EPS_) * (1.0f / TEMP_);
        const float sA  = accA[r] / fmaxf(sqrtf(n2a), EPS_) * (1.0f / TEMP_);
        if (trow != idxb) {
            sumV += __expf(sV);
            sumA += __expf(sA);
        }
    }
    // half-waves hold rows 0-7 / 8-15 respectively (identical across the half-wave)
    sumV += __shfl_xor(sumV, 16);
    sumA += __shfl_xor(sumA, 16);
    if (lane == 0) {
        atomicAdd(&negA[b], sumV);                   // a2v: video rows x a_anc
        atomicAdd(&negV[b], sumA);                   // v2a: audio rows x v_anc
    }
}

// ---------------------------------------------------------------------------
// Kernel 4: loss = 0.5 * sum_b(-2*pos + log(a2v) + log(v2a)) / B
// ---------------------------------------------------------------------------
__global__ __launch_bounds__(32) void finalize_kernel(const float* __restrict__ pos,
                                                      const float* __restrict__ negA,
                                                      const float* __restrict__ negV,
                                                      float* __restrict__ out) {
    float acc = 0.f;
    if (threadIdx.x < B_) {
        const int b = threadIdx.x;
        acc = -2.f * pos[b] + logf(negA[b]) + logf(negV[b]);
    }
    for (int off = 16; off > 0; off >>= 1) acc += __shfl_down(acc, off);
    if (threadIdx.x == 0) out[0] = acc * 0.5f / (float)B_;
}

// ---------------------------------------------------------------------------
extern "C" void kernel_launch(void* const* d_in, const int* in_sizes, int n_in,
                              void* d_out, int out_size, void* d_ws, size_t ws_size,
                              hipStream_t stream) {
    const float* video = (const float*)d_in[0];
    const float* audio = (const float*)d_in[1];
    const int*   mask  = (const int*)d_in[2];
    float*       out   = (float*)d_out;

    char* ws = (char*)d_ws;
    int*      idx   = (int*)(ws + 0);                 // 16 ints
    float*    pos   = (float*)(ws + 256);             // 16 floats
    float*    negA  = (float*)(ws + 512);             // 16 floats (a2v)
    float*    negV  = (float*)(ws + 768);             // 16 floats (v2a)
    _Float16* vanch = (_Float16*)(ws + 1024);         // 16*256 halfs (8 KB)
    _Float16* aanch = vanch + B_ * D_;                // 16*256 halfs (8 KB)

    argmax_kernel<<<B_, 256, 0, stream>>>(mask, idx);
    anchor_kernel<<<B_, 256, 0, stream>>>(video, audio, idx, pos, negA, negV,
                                          vanch, aanch);
    sim_kernel<<<dim3(T_ / 16 / 4, B_), 128, 0, stream>>>(video, audio, idx,
                                                          aanch, vanch, negA, negV);
    finalize_kernel<<<1, 32, 0, stream>>>(pos, negA, negV, out);
}
